// PAM_Module_3770981286552
// MI455X (gfx1250) — compile-verified, matrix operand order checked by reading, not run on hardware
//
#include <hip/hip_runtime.h>

// PAM attention, B=4, H=W=64, C=64 (N=4096).
// f16 WMMA for both big GEMMs; f32 softmax math; LDS tile sharing across
// 4 waves per workgroup, staged with global_load_async_to_lds_b128
// (ASYNCcnt double buffering).

typedef _Float16 v8h  __attribute__((ext_vector_type(8)));
typedef _Float16 v16h __attribute__((ext_vector_type(16)));
typedef float    v8f  __attribute__((ext_vector_type(8)));

#define NPIX 4096
#define CDIM 64
#define NB   4

#define WMMA_F16(A, B, C) \
  __builtin_amdgcn_wmma_f32_16x16x32_f16(false, (A), false, (B), (short)0, (C), false, false)

#define S_WAIT_ASYNCCNT(imm) asm volatile("s_wait_asynccnt " #imm ::: "memory")
#define S_WAIT_DSCNT0()      asm volatile("s_wait_dscnt 0x0" ::: "memory")

// 16B memory -> LDS, tracked by ASYNCcnt (one issue per active wave).
__device__ __forceinline__ void async_ld16(unsigned lds_byte, const void* gaddr) {
  asm volatile("global_load_async_to_lds_b128 %0, %1, off"
               :: "v"(lds_byte), "v"(gaddr) : "memory");
}
__device__ __forceinline__ unsigned lds_addr(const void* p) {
  return (unsigned)(unsigned long long)p;   // low 32 bits of generic = LDS offset
}

// 16x32 f16 A/B fragment: row = tile_row + lane%16, halves 0..7 at
// K = 8*(lane/16), halves 8..15 at K = 16 + 8*(lane/16).
__device__ __forceinline__ v16h load_frag(const _Float16* lo, const _Float16* hi) {
  v8h a = *(const v8h*)lo;
  v8h b = *(const v8h*)hi;
  v16h r;
#pragma unroll
  for (int i = 0; i < 8; ++i) { r[i] = a[i]; r[8 + i] = b[i]; }
  return r;
}

// ---------------------------------------------------------------------------
// Kernel 1: fused Q/K/V projection (1x1 conv). Q,K row-major f16; V stored
// transposed (c-major) so it is the A operand of out^T = V^T * P.
// ---------------------------------------------------------------------------
__global__ void __launch_bounds__(256)
proj_kernel(const float* __restrict__ x,
            const float* __restrict__ Wq, const float* __restrict__ bq,
            const float* __restrict__ Wk, const float* __restrict__ bk,
            const float* __restrict__ Wv, const float* __restrict__ bv,
            _Float16* __restrict__ qh, _Float16* __restrict__ kh,
            _Float16* __restrict__ vT) {
  __shared__ float xs[4][CDIM];
  const int tid = threadIdx.x;
  const int r = tid >> 6;
  const int f = tid & 63;
  const long long g = (long long)blockIdx.x * 4 + r;
  xs[r][f] = x[g * CDIM + f];
  __syncthreads();
  float aq = bq[f], ak = bk[f], av = bv[f];
#pragma unroll 8
  for (int c = 0; c < CDIM; ++c) {
    const float xv = xs[r][c];
    aq = fmaf(xv, Wq[c * CDIM + f], aq);
    ak = fmaf(xv, Wk[c * CDIM + f], ak);
    av = fmaf(xv, Wv[c * CDIM + f], av);
  }
  qh[g * CDIM + f] = (_Float16)aq;
  kh[g * CDIM + f] = (_Float16)ak;
  const int b = (int)(g >> 12);
  const int n = (int)(g & (NPIX - 1));
  vT[((long long)(b * CDIM + f)) * NPIX + n] = (_Float16)av;
}

// ---------------------------------------------------------------------------
// Kernel 2: streaming row stats. 4 waves/block, each wave one 16-row m-tile;
// the 32-row K tile per iteration is shared via LDS (async double buffer).
// ---------------------------------------------------------------------------
#define K2_STR  72                 // padded k-row stride in halves
#define K2_BUF  (32 * K2_STR)      // halves per buffer

__global__ void __launch_bounds__(128)
rowstats_kernel(const _Float16* __restrict__ qh, const _Float16* __restrict__ kh,
                float* __restrict__ rowmax, float* __restrict__ rowinv) {
  __shared__ __align__(16) _Float16 smem[2 * K2_BUF];
  const int tid  = threadIdx.x;
  const int lane = tid & 31;
  const int wv   = tid >> 5;
  const int h16  = lane >> 4;
  const int l16  = lane & 15;
  const int b    = blockIdx.x >> 6;
  const int m0   = (blockIdx.x & 63) * 64 + wv * 16;

  const _Float16* q = qh + (long long)b * NPIX * CDIM;
  const _Float16* k = kh + (long long)b * NPIX * CDIM;

  const _Float16* qrow = q + (m0 + l16) * CDIM;
  const v16h a0 = load_frag(qrow + h16 * 8,      qrow + 16 + h16 * 8);
  const v16h a1 = load_frag(qrow + 32 + h16 * 8, qrow + 48 + h16 * 8);

  const unsigned sbase = lds_addr(smem);
  // staging: 32 rows x 8 chunks of 16B; 2 chunks per thread
  const int ra = tid >> 3, ca = (tid & 7) * 8;
  const int rb = (tid + 128) >> 3, cb = ca;

  // prologue: iteration 0 into buffer 0
  async_ld16(sbase + (unsigned)(ra * K2_STR + ca) * 2, k + ra * CDIM + ca);
  async_ld16(sbase + (unsigned)(rb * K2_STR + cb) * 2, k + rb * CDIM + cb);

  float mx[8], sm[8];
#pragma unroll
  for (int i = 0; i < 8; ++i) { mx[i] = -3.0e38f; sm[i] = 0.0f; }

  for (int it = 0; it < NPIX / 32; ++it) {
    if (it + 1 < NPIX / 32) {
      const unsigned nb = sbase + (unsigned)(((it + 1) & 1) * K2_BUF) * 2;
      const _Float16* kb = k + (it + 1) * 32 * CDIM;
      async_ld16(nb + (unsigned)(ra * K2_STR + ca) * 2, kb + ra * CDIM + ca);
      async_ld16(nb + (unsigned)(rb * K2_STR + cb) * 2, kb + rb * CDIM + cb);
      S_WAIT_ASYNCCNT(0x2);
    } else {
      S_WAIT_ASYNCCNT(0x0);
    }
    __syncthreads();                            // buffer (it&1) staged
    const _Float16* ks = smem + (it & 1) * K2_BUF;
    const _Float16* kr0 = ks + l16 * K2_STR;
    const _Float16* kr1 = ks + (16 + l16) * K2_STR;
    const v16h b00 = load_frag(kr0 + h16 * 8,      kr0 + 16 + h16 * 8);
    const v16h b01 = load_frag(kr0 + 32 + h16 * 8, kr0 + 48 + h16 * 8);
    const v16h b10 = load_frag(kr1 + h16 * 8,      kr1 + 16 + h16 * 8);
    const v16h b11 = load_frag(kr1 + 32 + h16 * 8, kr1 + 48 + h16 * 8);
    S_WAIT_DSCNT0();
    __syncthreads();                            // safe to overwrite buffer
    v8f e0 = {}, e1 = {};
    e0 = WMMA_F16(a0, b00, e0);
    e0 = WMMA_F16(a1, b01, e0);
    e1 = WMMA_F16(a0, b10, e1);
    e1 = WMMA_F16(a1, b11, e1);
#pragma unroll
    for (int i = 0; i < 8; ++i) {
      float v = e0[i];
      float nm = fmaxf(mx[i], v);
      sm[i] = sm[i] * __expf(mx[i] - nm) + __expf(v - nm);
      mx[i] = nm;
      v = e1[i];
      nm = fmaxf(mx[i], v);
      sm[i] = sm[i] * __expf(mx[i] - nm) + __expf(v - nm);
      mx[i] = nm;
    }
  }
  // merge across the 16 lanes sharing each row m
#pragma unroll
  for (int i = 0; i < 8; ++i) {
    float m = mx[i], s = sm[i];
#pragma unroll
    for (int off = 8; off >= 1; off >>= 1) {
      const float om = __shfl_xor(m, off, 32);
      const float os = __shfl_xor(s, off, 32);
      const float nm = fmaxf(m, om);
      s = s * __expf(m - nm) + os * __expf(om - nm);
      m = nm;
    }
    mx[i] = m; sm[i] = s;
  }
  if (l16 == 0) {
    const int base = b * NPIX + m0 + h16 * 8;
#pragma unroll
    for (int i = 0; i < 8; ++i) {
      rowmax[base + i] = mx[i];
      rowinv[base + i] = 1.0f / sm[i];
    }
  }
}

// ---------------------------------------------------------------------------
// Kernel 3: out^T = V^T * softmax(E). 4 waves/block, each wave one 16-col
// n-tile; the per-iteration Q tile (32x64), V^T tile (64x32) and stats are
// shared via LDS (async double buffer, 5 async issues per wave per step).
// Energy D-tile register layout == second WMMA's B-operand layout, so P is
// fed through registers with no transpose.
// ---------------------------------------------------------------------------
#define K3_QSTR 72                      // padded q-row stride (halves)
#define K3_VSTR 40                      // padded vT-row stride (halves)
#define K3_QSZ  (32 * K3_QSTR)          // 2304 halves
#define K3_VSZ  (64 * K3_VSTR)          // 2560 halves
#define K3_SSZ  128                     // 64 floats (max[32] + inv[32])
#define K3_BUF  (K3_QSZ + K3_VSZ + K3_SSZ)

__device__ __forceinline__ void k3_stage(unsigned sbase, int bb, int m0, int tid,
                                         const _Float16* __restrict__ q,
                                         const _Float16* __restrict__ vt,
                                         const float* __restrict__ mxp,
                                         const float* __restrict__ ivp) {
  const unsigned base = sbase + (unsigned)(bb * K3_BUF) * 2;
  // Q tile: 32 rows x 8 chunks
  {
    int idx = tid, r = idx >> 3, c = (idx & 7) * 8;
    async_ld16(base + (unsigned)(r * K3_QSTR + c) * 2, q + (m0 + r) * CDIM + c);
    idx = tid + 128; r = idx >> 3; c = (idx & 7) * 8;
    async_ld16(base + (unsigned)(r * K3_QSTR + c) * 2, q + (m0 + r) * CDIM + c);
  }
  // V^T tile: 64 rows x 4 chunks
  const unsigned vb = base + (unsigned)K3_QSZ * 2;
  {
    int idx = tid, r = idx >> 2, c = (idx & 3) * 8;
    async_ld16(vb + (unsigned)(r * K3_VSTR + c) * 2, vt + r * NPIX + m0 + c);
    idx = tid + 128; r = idx >> 2; c = (idx & 3) * 8;
    async_ld16(vb + (unsigned)(r * K3_VSTR + c) * 2, vt + r * NPIX + m0 + c);
  }
  // stats: lanes 0..15 of every wave (redundant, keeps per-wave count fixed)
  const unsigned sb = base + (unsigned)(K3_QSZ + K3_VSZ) * 2;
  const int l = tid & 31;
  const int sl = l & 15;
  const float* g = (sl < 8) ? (mxp + m0 + sl * 4) : (ivp + m0 + (sl - 8) * 4);
  if (l < 16) async_ld16(sb + (unsigned)sl * 16, g);
}

__global__ void __launch_bounds__(128)
attnout_kernel(const _Float16* __restrict__ qh, const _Float16* __restrict__ kh,
               const _Float16* __restrict__ vT,
               const float* __restrict__ rowmax, const float* __restrict__ rowinv,
               const float* __restrict__ x, const float* __restrict__ gamma,
               float* __restrict__ out) {
  __shared__ __align__(16) _Float16 smem[2 * K3_BUF];
  const int tid  = threadIdx.x;
  const int lane = tid & 31;
  const int wv   = tid >> 5;
  const int h16  = lane >> 4;
  const int l16  = lane & 15;
  const int b    = blockIdx.x >> 6;
  const int n0   = (blockIdx.x & 63) * 64 + wv * 16;

  const _Float16* q  = qh + (long long)b * NPIX * CDIM;
  const _Float16* k  = kh + (long long)b * NPIX * CDIM;
  const _Float16* vt = vT + (long long)b * CDIM * NPIX;
  const float* mx_base = rowmax + b * NPIX;
  const float* iv_base = rowinv + b * NPIX;

  // B operand of the energy WMMAs: k rows n0..n0+15 (fixed per wave)
  const _Float16* krow = k + (n0 + l16) * CDIM;
  const v16h bk0 = load_frag(krow + h16 * 8,      krow + 16 + h16 * 8);
  const v16h bk1 = load_frag(krow + 32 + h16 * 8, krow + 48 + h16 * 8);

  const unsigned sbase = lds_addr(smem);
  k3_stage(sbase, 0, 0, tid, q, vt, mx_base, iv_base);

  v8f acc0 = {}, acc1 = {}, acc2 = {}, acc3 = {};

  for (int mt = 0; mt < NPIX / 32; ++mt) {
    const int m0 = mt * 32;
    if (mt + 1 < NPIX / 32) {
      k3_stage(sbase, (mt + 1) & 1, m0 + 32, tid, q, vt, mx_base, iv_base);
      S_WAIT_ASYNCCNT(0x5);
    } else {
      S_WAIT_ASYNCCNT(0x0);
    }
    __syncthreads();                      // buffer (mt&1) staged
    const _Float16* sb = smem + (mt & 1) * K3_BUF;
    const _Float16* qs = sb;
    const _Float16* vs = sb + K3_QSZ;
    const float*    st = (const float*)(sb + K3_QSZ + K3_VSZ);

    const _Float16* q0 = qs + l16 * K3_QSTR;
    const _Float16* q1 = qs + (16 + l16) * K3_QSTR;
    const v16h a0 = load_frag(q0 + h16 * 8,      q0 + 16 + h16 * 8);
    const v16h a1 = load_frag(q0 + 32 + h16 * 8, q0 + 48 + h16 * 8);
    const v16h a2 = load_frag(q1 + h16 * 8,      q1 + 16 + h16 * 8);
    const v16h a3 = load_frag(q1 + 32 + h16 * 8, q1 + 48 + h16 * 8);

    const _Float16* v0 = vs + (0 * 16 + l16) * K3_VSTR;
    const _Float16* v1 = vs + (1 * 16 + l16) * K3_VSTR;
    const _Float16* v2 = vs + (2 * 16 + l16) * K3_VSTR;
    const _Float16* v3 = vs + (3 * 16 + l16) * K3_VSTR;
    const v16h va0 = load_frag(v0 + h16 * 8, v0 + 16 + h16 * 8);
    const v16h va1 = load_frag(v1 + h16 * 8, v1 + 16 + h16 * 8);
    const v16h va2 = load_frag(v2 + h16 * 8, v2 + 16 + h16 * 8);
    const v16h va3 = load_frag(v3 + h16 * 8, v3 + 16 + h16 * 8);

    const float* lm = st;
    const float* li = st + 32;
    const float4 mA = *(const float4*)(lm + h16 * 8);
    const float4 mB = *(const float4*)(lm + h16 * 8 + 4);
    const float4 mC = *(const float4*)(lm + h16 * 8 + 16);
    const float4 mD = *(const float4*)(lm + h16 * 8 + 20);
    const float4 iA = *(const float4*)(li + h16 * 8);
    const float4 iB = *(const float4*)(li + h16 * 8 + 4);
    const float4 iC = *(const float4*)(li + h16 * 8 + 16);
    const float4 iD = *(const float4*)(li + h16 * 8 + 20);
    S_WAIT_DSCNT0();
    __syncthreads();                      // safe to overwrite buffer

    v8f e0 = {}, e1 = {};
    e0 = WMMA_F16(a0, bk0, e0);
    e0 = WMMA_F16(a1, bk1, e0);
    e1 = WMMA_F16(a2, bk0, e1);
    e1 = WMMA_F16(a3, bk1, e1);

    const float mx0[8] = {mA.x, mA.y, mA.z, mA.w, mB.x, mB.y, mB.z, mB.w};
    const float mx1[8] = {mC.x, mC.y, mC.z, mC.w, mD.x, mD.y, mD.z, mD.w};
    const float iv0[8] = {iA.x, iA.y, iA.z, iA.w, iB.x, iB.y, iB.z, iB.w};
    const float iv1[8] = {iC.x, iC.y, iC.z, iC.w, iD.x, iD.y, iD.z, iD.w};
    v16h pb;
#pragma unroll
    for (int r = 0; r < 8; ++r) {
      pb[r]     = (_Float16)(__expf(e0[r] - mx0[r]) * iv0[r]);
      pb[8 + r] = (_Float16)(__expf(e1[r] - mx1[r]) * iv1[r]);
    }
    acc0 = WMMA_F16(va0, pb, acc0);
    acc1 = WMMA_F16(va1, pb, acc1);
    acc2 = WMMA_F16(va2, pb, acc2);
    acc3 = WMMA_F16(va3, pb, acc3);
  }

  // Epilogue: D element (c = ct*16 + r + 8*h16, n = n0 + l16)
  const long long row = (long long)(b * NPIX + n0 + l16) * CDIM;
  const float* xr = x + row;
  float* outr = out + row;
  const float g = gamma[0];
  const v8f accs[4] = {acc0, acc1, acc2, acc3};
#pragma unroll
  for (int ct = 0; ct < 4; ++ct) {
    const int base = ct * 16 + h16 * 8;
    float4 oA, oB;
    oA.x = fmaf(g, accs[ct][0], xr[base + 0]);
    oA.y = fmaf(g, accs[ct][1], xr[base + 1]);
    oA.z = fmaf(g, accs[ct][2], xr[base + 2]);
    oA.w = fmaf(g, accs[ct][3], xr[base + 3]);
    oB.x = fmaf(g, accs[ct][4], xr[base + 4]);
    oB.y = fmaf(g, accs[ct][5], xr[base + 5]);
    oB.z = fmaf(g, accs[ct][6], xr[base + 6]);
    oB.w = fmaf(g, accs[ct][7], xr[base + 7]);
    *(float4*)(outr + base)     = oA;
    *(float4*)(outr + base + 4) = oB;
  }
}

// ---------------------------------------------------------------------------
extern "C" void kernel_launch(void* const* d_in, const int* in_sizes, int n_in,
                              void* d_out, int out_size, void* d_ws, size_t ws_size,
                              hipStream_t stream) {
  (void)in_sizes; (void)n_in; (void)out_size; (void)ws_size;
  const float* x     = (const float*)d_in[0];
  const float* Wq    = (const float*)d_in[1];
  const float* bq    = (const float*)d_in[2];
  const float* Wk    = (const float*)d_in[3];
  const float* bk    = (const float*)d_in[4];
  const float* Wv    = (const float*)d_in[5];
  const float* bv    = (const float*)d_in[6];
  const float* gamma = (const float*)d_in[7];
  float* out = (float*)d_out;

  // workspace: q(2MB) k(2MB) vT(2MB) rowmax(64KB) rowinv(64KB)
  _Float16* qh = (_Float16*)d_ws;
  _Float16* kh = qh + (size_t)NB * NPIX * CDIM;
  _Float16* vT = kh + (size_t)NB * NPIX * CDIM;
  float* rowmax = (float*)(vT + (size_t)NB * NPIX * CDIM);
  float* rowinv = rowmax + (size_t)NB * NPIX;

  proj_kernel<<<NB * NPIX / 4, 256, 0, stream>>>(x, Wq, bq, Wk, bk, Wv, bv, qh, kh, vT);
  rowstats_kernel<<<NB * 64, 128, 0, stream>>>(qh, kh, rowmax, rowinv);
  attnout_kernel<<<NB * 64, 128, 0, stream>>>(qh, kh, vT, rowmax, rowinv, x, gamma, out);
}
